// CPMAntAttention_3066606649481
// MI455X (gfx1250) — matrix operand order repaired and checked by reading, not compiled, not run on hardware
//
#include <hip/hip_runtime.h>
#include <hip/hip_bf16.h>
#include <math.h>
#include <stdint.h>

typedef __attribute__((ext_vector_type(16))) _Float16 v16h;
typedef __attribute__((ext_vector_type(8)))  _Float16 v8h;
typedef __attribute__((ext_vector_type(8)))  float    v8f;

#define BQ   2
#define LQQ  2048
#define LKK  2048
#define DMM  2048
#define HH   16
#define DHH  128

// ---- fragment loaders (CDNA5 wave32 WMMA layouts) ----------------------
// A-matrix 16x32 f16: lane L -> M = L%16 ; halves: K = 8*(L/16) + {0..7, 16..23}
__device__ __forceinline__ v16h load_a_frag(const _Float16* row, int k0, int g) {
  v16h a;
  *(v8h*)&a       = *(const v8h*)(row + k0 + 8 * g);
  *((v8h*)&a + 1) = *(const v8h*)(row + k0 + 8 * g + 16);
  return a;
}
// B-matrix 32x16 f16: lane L -> N = L%16 ; halves: K = 8*(L/16) + {0..7, 16..23}
__device__ __forceinline__ v16h load_b_frag(const _Float16* row, int k0, int g) {
  v16h b;
  *(v8h*)&b       = *(const v8h*)(row + k0 + 8 * g);
  *((v8h*)&b + 1) = *(const v8h*)(row + k0 + 8 * g + 16);
  return b;
}

// per-lane async global->LDS copy of 16 bytes (ASYNCcnt-tracked DMA)
__device__ __forceinline__ void async_cp16(uint32_t lds_addr, const void* gaddr) {
  asm volatile("global_load_async_to_lds_b128 %0, %1, off"
               :: "v"(lds_addr), "v"((uint64_t)(size_t)gaddr)
               : "memory");
}

// ---- f32 -> f16 conversion with scale ----------------------------------
__global__ void cvt_f32_f16(const float* __restrict__ in, _Float16* __restrict__ out,
                            float scale, int n) {
  int i = blockIdx.x * blockDim.x + threadIdx.x;
  if (i < n) out[i] = (_Float16)(in[i] * scale);
}

// ---- NT GEMM: C[M,N] = A[M,K] * W[N,K]^T -------------------------------
// LDS double-buffered via async global->LDS DMA; 4 waves, 128x128 block tile.
// MODE 0: store f16 to (B,H,Lseq,DH)   [q/k layout]
// MODE 1: store f16 to (B,H,DH,Lseq)   [v transposed layout]
// MODE 3: store f32 row-major M x N    [final output]
template <int MODE>
__global__ __launch_bounds__(128) void gemm_nt(const _Float16* __restrict__ A,
                                               const _Float16* __restrict__ W,
                                               void* __restrict__ Cout,
                                               int M, int N, int K, int Lseq) {
  __shared__ __align__(16) _Float16 sA[2][128 * 32];
  __shared__ __align__(16) _Float16 sW[2][128 * 32];
  const int tid  = threadIdx.x;
  const int lane = tid & 31;
  const int wave = tid >> 5;
  const int g  = lane >> 4;
  const int ln = lane & 15;
  const int mblk = blockIdx.y * 128;
  const int nblk = blockIdx.x * 128;
  const int m0 = (wave >> 1) * 64;  // within block
  const int n0 = (wave & 1) * 64;

  // stage one 128x32 tile of A and of W into LDS buffer `buf`
  // (each of 128 threads owns one row: 64B = 4 async b128; 8 asyncs/thread)
  auto stage = [&](int buf, int k0) {
    const _Float16* arow = A + (size_t)(mblk + tid) * K + k0;
    const _Float16* wrow = W + (size_t)(nblk + tid) * K + k0;
    uint32_t la = (uint32_t)(size_t)(&sA[buf][tid * 32]);
    uint32_t lw = (uint32_t)(size_t)(&sW[buf][tid * 32]);
#pragma unroll
    for (int c = 0; c < 4; c++) {
      async_cp16(la + c * 16, arow + c * 8);
      async_cp16(lw + c * 16, wrow + c * 8);
    }
  };

  v8f z = {};
  v8f acc[4][4];
#pragma unroll
  for (int i = 0; i < 4; i++)
#pragma unroll
    for (int j = 0; j < 4; j++) acc[i][j] = z;

  const int nk = K / 32;
  stage(0, 0);
  for (int kk = 0; kk < nk; kk++) {
    const int buf = kk & 1;
    if (kk + 1 < nk) {
      stage(buf ^ 1, (kk + 1) * 32);
      // oldest stage (buf) complete when <= 8 (this wave's new stage) pending
      asm volatile("s_wait_asynccnt 0x8" ::: "memory");
    } else {
      asm volatile("s_wait_asynccnt 0x0" ::: "memory");
    }
    __syncthreads();  // all waves' DMA for `buf` visible

    v16h aF[4];
#pragma unroll
    for (int mt = 0; mt < 4; mt++)
      aF[mt] = load_a_frag(&sA[buf][(m0 + mt * 16 + ln) * 32], 0, g);
#pragma unroll
    for (int nt = 0; nt < 4; nt++) {
      v16h bF = load_b_frag(&sW[buf][(n0 + nt * 16 + ln) * 32], 0, g);
#pragma unroll
      for (int mt = 0; mt < 4; mt++)
        acc[mt][nt] = __builtin_amdgcn_wmma_f32_16x16x32_f16(
            false, aF[mt], false, bF, (short)0, acc[mt][nt], false, false);
    }
    __syncthreads();  // reads done before buf is re-staged
  }

#pragma unroll
  for (int mt = 0; mt < 4; mt++)
#pragma unroll
    for (int nt = 0; nt < 4; nt++)
#pragma unroll
      for (int r = 0; r < 8; r++) {
        int m = mblk + m0 + mt * 16 + r + 8 * g;
        int n = nblk + n0 + nt * 16 + ln;
        float val = acc[mt][nt][r];
        if (MODE == 0) {
          int b = m / Lseq, l = m % Lseq, hh = n / DHH, d = n % DHH;
          ((_Float16*)Cout)[((((size_t)b * HH + hh) * Lseq + l) * DHH) + d] = (_Float16)val;
        } else if (MODE == 1) {
          int b = m / Lseq, l = m % Lseq, hh = n / DHH, d = n % DHH;
          ((_Float16*)Cout)[((((size_t)b * HH + hh) * DHH + d) * Lseq) + l] = (_Float16)val;
        } else {
          ((float*)Cout)[(size_t)m * N + n] = val;
        }
      }
}

// ---- flash attention: per wave, 16 q rows x full DH, kv chunk = 64 -----
__global__ __launch_bounds__(256) void flash_attn(
    const _Float16* __restrict__ Q,   // (B,H,LQ,DH)
    const _Float16* __restrict__ Km,  // (B,H,LK,DH)
    const _Float16* __restrict__ Vt,  // (B,H,DH,LK)
    const float* __restrict__ bias,   // (H,LQ,LK)
    const unsigned char* __restrict__ mask,  // (B,LQ,LK)
    _Float16* __restrict__ Out) {            // (B,LQ,H*DH)
  __shared__ __align__(16) _Float16 lds[8][16 * 64];
  const int lane = threadIdx.x & 31;
  const int wave = threadIdx.x >> 5;
  const int g = lane >> 4, ln = lane & 15;
  const int b = blockIdx.x / HH, h = blockIdx.x % HH;
  const int q0 = blockIdx.y * 128 + wave * 16;

  const _Float16* qb = Q  + ((size_t)b * HH + h) * LQQ * DHH;
  const _Float16* kb = Km + ((size_t)b * HH + h) * LKK * DHH;
  const _Float16* vb = Vt + ((size_t)b * HH + h) * DHH * LKK;
  const float* pbias = bias + (size_t)h * LQQ * LKK;
  const unsigned char* pmask = mask + (size_t)b * LQQ * LKK;

  v16h aq[4];
#pragma unroll
  for (int f = 0; f < 4; f++)
    aq[f] = load_a_frag(qb + (size_t)(q0 + ln) * DHH, f * 32, g);

  v8f z = {};
  v8f o[8];
#pragma unroll
  for (int t = 0; t < 8; t++) o[t] = z;
  float mrow[8], lrow[8];
#pragma unroll
  for (int r = 0; r < 8; r++) { mrow[r] = -3.0e38f; lrow[r] = 0.f; }

  for (int kv0 = 0; kv0 < LKK; kv0 += 64) {
    // S = Q*K^T (scale folded into wq): four 16-wide N tiles
    v8f s[4];
#pragma unroll
    for (int j = 0; j < 4; j++) {
      v8f a = z;
      const _Float16* krow = kb + (size_t)(kv0 + j * 16 + ln) * DHH;
#pragma unroll
      for (int f = 0; f < 4; f++) {
        v16h bk = load_b_frag(krow, f * 32, g);
        a = __builtin_amdgcn_wmma_f32_16x16x32_f16(false, aq[f], false, bk,
                                                   (short)0, a, false, false);
      }
      s[j] = a;
    }
    // bias + mask in C-layout
#pragma unroll
    for (int j = 0; j < 4; j++)
#pragma unroll
      for (int r = 0; r < 8; r++) {
        int row = q0 + r + 8 * g;
        int col = kv0 + j * 16 + ln;
        float sv = s[j][r] + pbias[(size_t)row * LKK + col];
        if (!pmask[(size_t)row * LKK + col]) sv = -1.0e30f;
        s[j][r] = sv;
      }
    // online softmax (row = r + 8*g lives in one 16-lane half)
    float corr[8];
#pragma unroll
    for (int r = 0; r < 8; r++) {
      float t = fmaxf(fmaxf(s[0][r], s[1][r]), fmaxf(s[2][r], s[3][r]));
#pragma unroll
      for (int off = 1; off < 16; off <<= 1) t = fmaxf(t, __shfl_xor(t, off, 16));
      float mn = fmaxf(mrow[r], t);
      corr[r] = __expf(mrow[r] - mn);
      mrow[r] = mn;
      float rs = 0.f;
#pragma unroll
      for (int j = 0; j < 4; j++) {
        float p = __expf(s[j][r] - mn);
        s[j][r] = p;
        rs += p;
      }
#pragma unroll
      for (int off = 1; off < 16; off <<= 1) rs += __shfl_xor(rs, off, 16);
      lrow[r] = lrow[r] * corr[r] + rs;
    }
#pragma unroll
    for (int t = 0; t < 8; t++)
#pragma unroll
      for (int r = 0; r < 8; r++) o[t][r] *= corr[r];

    // stage P (16x64 f16) through LDS to convert C-layout -> A-layout
    _Float16* pl = &lds[wave][0];
#pragma unroll
    for (int r = 0; r < 8; r++) {
      int row = r + 8 * g;
#pragma unroll
      for (int j = 0; j < 4; j++) pl[row * 64 + j * 16 + ln] = (_Float16)s[j][r];
    }
    asm volatile("" ::: "memory");  // LDS is in-order within a wave
    v16h pA0, pA1;
    *(v8h*)&pA0       = *(const v8h*)(pl + ln * 64 + 8 * g);
    *((v8h*)&pA0 + 1) = *(const v8h*)(pl + ln * 64 + 8 * g + 16);
    *(v8h*)&pA1       = *(const v8h*)(pl + ln * 64 + 32 + 8 * g);
    *((v8h*)&pA1 + 1) = *(const v8h*)(pl + ln * 64 + 32 + 8 * g + 16);

    // O += P * V  (V pre-transposed: rows are d, contiguous over kv)
#pragma unroll
    for (int t = 0; t < 8; t++) {
      const _Float16* vrow = vb + (size_t)(t * 16 + ln) * LKK + kv0;
      v16h bv0, bv1;
      *(v8h*)&bv0       = *(const v8h*)(vrow + 8 * g);
      *((v8h*)&bv0 + 1) = *(const v8h*)(vrow + 8 * g + 16);
      *(v8h*)&bv1       = *(const v8h*)(vrow + 32 + 8 * g);
      *((v8h*)&bv1 + 1) = *(const v8h*)(vrow + 32 + 8 * g + 16);
      o[t] = __builtin_amdgcn_wmma_f32_16x16x32_f16(false, pA0, false, bv0,
                                                    (short)0, o[t], false, false);
      o[t] = __builtin_amdgcn_wmma_f32_16x16x32_f16(false, pA1, false, bv1,
                                                    (short)0, o[t], false, false);
    }
  }

  // normalize + store (B,LQ,H*DH) f16
#pragma unroll
  for (int r = 0; r < 8; r++) {
    float inv = 1.0f / lrow[r];
    int row = q0 + r + 8 * g;
    _Float16* orow = Out + ((size_t)b * LQQ + row) * (HH * DHH) + h * DHH;
#pragma unroll
    for (int t = 0; t < 8; t++) orow[t * 16 + ln] = (_Float16)(o[t][r] * inv);
  }
}

extern "C" void kernel_launch(void* const* d_in, const int* in_sizes, int n_in,
                              void* d_out, int out_size, void* d_ws, size_t ws_size,
                              hipStream_t stream) {
  const float* hq  = (const float*)d_in[0];
  const float* hkv = (const float*)d_in[1];
  const unsigned char* mask = (const unsigned char*)d_in[2];
  const float* bias = (const float*)d_in[3];
  const float* wq = (const float*)d_in[4];
  const float* wk = (const float*)d_in[5];
  const float* wv = (const float*)d_in[6];
  const float* wo = (const float*)d_in[7];

  char* ws = (char*)d_ws;
  size_t off = 0;
  auto alloc16 = [&](size_t nhalves) {
    _Float16* p = (_Float16*)(ws + off);
    off += nhalves * sizeof(_Float16);
    return p;
  };
  const size_t NHID = (size_t)BQ * LQQ * DMM;   // 8M
  const size_t NW   = (size_t)DMM * DMM;        // 4M
  _Float16* hq16  = alloc16(NHID);
  _Float16* hkv16 = alloc16(NHID);
  _Float16* wq16  = alloc16(NW);
  _Float16* wk16  = alloc16(NW);
  _Float16* wv16  = alloc16(NW);
  _Float16* wo16  = alloc16(NW);
  _Float16* q16   = alloc16(NHID);  // (B,H,LQ,DH)
  _Float16* k16   = alloc16(NHID);  // (B,H,LK,DH)
  _Float16* vt16  = alloc16(NHID);  // (B,H,DH,LK)
  _Float16* ao16  = alloc16(NHID);  // (B,LQ,H*DH)

  const float sProj = 1.0f / sqrtf((float)DMM);
  const float sQ    = sProj / sqrtf((float)DHH);   // fold 1/sqrt(DH) into wq
  const float sO    = 1.0f / sqrtf((float)(HH * DHH));

  const int T = 256;
  cvt_f32_f16<<<(int)(NHID / T), T, 0, stream>>>(hq,  hq16,  1.0f,  (int)NHID);
  cvt_f32_f16<<<(int)(NHID / T), T, 0, stream>>>(hkv, hkv16, 1.0f,  (int)NHID);
  cvt_f32_f16<<<(int)(NW / T),   T, 0, stream>>>(wq,  wq16,  sQ,    (int)NW);
  cvt_f32_f16<<<(int)(NW / T),   T, 0, stream>>>(wk,  wk16,  sProj, (int)NW);
  cvt_f32_f16<<<(int)(NW / T),   T, 0, stream>>>(wv,  wv16,  sProj, (int)NW);
  cvt_f32_f16<<<(int)(NW / T),   T, 0, stream>>>(wo,  wo16,  sO,    (int)NW);

  const int M = BQ * LQQ;  // 4096
  dim3 gg(DMM / 128, M / 128);
  gemm_nt<0><<<gg, 128, 0, stream>>>(hq16,  wq16, q16,  M, DMM, DMM, LQQ);
  gemm_nt<0><<<gg, 128, 0, stream>>>(hkv16, wk16, k16,  M, DMM, DMM, LKK);
  gemm_nt<1><<<gg, 128, 0, stream>>>(hkv16, wv16, vt16, M, DMM, DMM, LKK);

  flash_attn<<<dim3(BQ * HH, LQQ / 128), 256, 0, stream>>>(q16, k16, vt16, bias,
                                                           mask, ao16);

  gemm_nt<3><<<gg, 128, 0, stream>>>(ao16, wo16, d_out, M, DMM, DMM, LQQ);
}
